// OnnxGaussianUpsampling_20761871909293
// MI455X (gfx1250) — matrix operand order repaired and checked by reading, not compiled, not run on hardware
//
#include <hip/hip_runtime.h>

typedef __attribute__((ext_vector_type(16))) _Float16 v16h;
typedef __attribute__((ext_vector_type(8)))  _Float16 v8h;
typedef __attribute__((ext_vector_type(8)))  float    v8f;

#define T_TEXT 1024
#define CHN    384
#define NB     4
#define DELTA  0.1f

// ---------------------------------------------------------------------------
// Kernel 1: Gaussian centers  c[b][t] = cumsum(ds)[t] - ds[t]/2   (f32)
// ---------------------------------------------------------------------------
__global__ void gup_centers_kernel(const int* __restrict__ ds,
                                   float* __restrict__ cbuf) {
  const int b = blockIdx.x;
  if (threadIdx.x == 0) {
    float run = 0.0f;
    for (int t = 0; t < T_TEXT; ++t) {
      const float d = (float)ds[b * T_TEXT + t];
      run += d;
      cbuf[b * T_TEXT + t] = run - 0.5f * d;
    }
  }
}

// ---------------------------------------------------------------------------
// Kernel 2: hs [B][T][C] f32  ->  hsT [B][C][T] f16  (WMMA B-operand layout:
// per output column, K runs contiguously). 3 MB -> stays in 192 MB L2.
// ---------------------------------------------------------------------------
__global__ void gup_cvt_kernel(const float* __restrict__ hs,
                               _Float16* __restrict__ hsT) {
  const int idx = blockIdx.x * 256 + threadIdx.x;
  if (idx >= NB * T_TEXT * CHN) return;
  const int b  = idx / (T_TEXT * CHN);
  const int r  = idx % (T_TEXT * CHN);
  const int t  = r / CHN;
  const int ch = r % CHN;
  hsT[((size_t)b * CHN + ch) * T_TEXT + t] = (_Float16)hs[idx];
}

// ---------------------------------------------------------------------------
// Kernel 3: fused softmax + WMMA matmul.
// Block = 256 threads (8 wave32) owns 32 frames x all 384 channels:
// 2 M-tiles x 24 N-tiles; each wave does 2x3 tiles so every B fragment
// feeds two v_wmma ops (halves L2 read traffic vs 1 M-tile blocking).
// ---------------------------------------------------------------------------
__global__ __launch_bounds__(256)
void gup_main_kernel(const float* __restrict__ cbuf,
                     const unsigned char* __restrict__ dmask,
                     const _Float16* __restrict__ hsT,
                     float* __restrict__ out,
                     int T_feats) {
  __shared__ __attribute__((aligned(32))) _Float16 Pa[32][T_TEXT]; // 64 KB p_attn tile (f16)
  __shared__ float cl[T_TEXT];    // centers
  __shared__ float ml[T_TEXT];    // mask as 0/1 float
  __shared__ float red[32][8];    // per-frame partial sums
  __shared__ float rinv[32];      // 1/rowsum

  const int tid = threadIdx.x;
  const int b   = blockIdx.y;
  const int f0  = blockIdx.x * 32;

  // ---- stage centers + mask into LDS -------------------------------------
  for (int i = tid; i < T_TEXT; i += 256) {
    cl[i] = cbuf[b * T_TEXT + i];
    ml[i] = dmask[b * T_TEXT + i] ? 1.0f : 0.0f;
  }
  __syncthreads();

  // ---- softmax numerators (no max-sub needed: energy<=0, sum >= e^-9) ----
  const int fr  = tid >> 3;          // frame within tile: 0..31
  const int sub = tid & 7;           // 8 threads per frame, 128 t's each
  const float fcoord = (float)(f0 + fr);
  const int tbase = sub * 128;
  float s = 0.0f;
  #pragma unroll 4
  for (int i = 0; i < 128; ++i) {
    const int t = tbase + i;
    const float d = fcoord - cl[t];
    const float w = ml[t] * __expf(-DELTA * d * d);
    Pa[fr][t] = (_Float16)w;
    s += w;
  }
  red[fr][sub] = s;
  __syncthreads();
  if (sub < 4) red[fr][sub] += red[fr][sub + 4];
  __syncthreads();
  if (sub < 2) red[fr][sub] += red[fr][sub + 2];
  __syncthreads();
  if (sub == 0) {
    const float tot = red[fr][0] + red[fr][1];
    rinv[fr] = (tot > 0.0f) ? (1.0f / tot) : 0.0f;
  }
  __syncthreads();
  {
    const float inv = rinv[fr];
    #pragma unroll 4
    for (int i = 0; i < 128; ++i) {
      const int t = tbase + i;
      Pa[fr][t] = (_Float16)((float)Pa[fr][t] * inv);
    }
  }
  __syncthreads();

  // ---- WMMA phase: wave handles 2 M-tiles x 3 N-tiles --------------------
  const int wave = tid >> 5;         // 0..7
  const int lane = tid & 31;
  const int n = lane & 15;           // A: row M / B,D: column N
  const int g = lane >> 4;           // lane-group (K-half / M-half select)

  const int ct0 = wave, ct1 = wave + 8, ct2 = wave + 16;
  const _Float16* hb  = hsT + (size_t)b * CHN * T_TEXT;
  const _Float16* bp0 = hb + (size_t)(ct0 * 16 + n) * T_TEXT + g * 16;
  const _Float16* bp1 = hb + (size_t)(ct1 * 16 + n) * T_TEXT + g * 16;
  const _Float16* bp2 = hb + (size_t)(ct2 * 16 + n) * T_TEXT + g * 16;

  v8f acc00 = {}; v8f acc01 = {}; v8f acc02 = {};   // M-tile 0 x {ct0,ct1,ct2}
  v8f acc10 = {}; v8f acc11 = {}; v8f acc12 = {};   // M-tile 1 x {ct0,ct1,ct2}

  for (int kb = 0; kb < T_TEXT / 32; ++kb) {
    const int k0 = kb * 32;
    // A fragments (contiguous K order, as in B): dwords 0-3 = K[g*16 .. g*16+7],
    // dwords 4-7 = K[g*16+8 .. g*16+15] -> two aligned ds_load_b128 each.
    const v8h a0lo = *(const v8h*)&Pa[n][k0 + g * 16];
    const v8h a0hi = *(const v8h*)&Pa[n][k0 + g * 16 + 8];
    const v8h a1lo = *(const v8h*)&Pa[16 + n][k0 + g * 16];
    const v8h a1hi = *(const v8h*)&Pa[16 + n][k0 + g * 16 + 8];
    v16h a0, a1;
    #pragma unroll
    for (int i = 0; i < 8; ++i) {
      a0[i] = a0lo[i]; a0[i + 8] = a0hi[i];
      a1[i] = a1lo[i]; a1[i + 8] = a1hi[i];
    }

    // B fragments (same contiguous K order as A): column n, 16 contiguous K values.
    const v16h vb0 = *(const v16h*)(bp0 + k0);
    const v16h vb1 = *(const v16h*)(bp1 + k0);
    const v16h vb2 = *(const v16h*)(bp2 + k0);

    acc00 = __builtin_amdgcn_wmma_f32_16x16x32_f16(false, a0, false, vb0,
                                                   (short)0, acc00, false, false);
    acc10 = __builtin_amdgcn_wmma_f32_16x16x32_f16(false, a1, false, vb0,
                                                   (short)0, acc10, false, false);
    acc01 = __builtin_amdgcn_wmma_f32_16x16x32_f16(false, a0, false, vb1,
                                                   (short)0, acc01, false, false);
    acc11 = __builtin_amdgcn_wmma_f32_16x16x32_f16(false, a1, false, vb1,
                                                   (short)0, acc11, false, false);
    acc02 = __builtin_amdgcn_wmma_f32_16x16x32_f16(false, a0, false, vb2,
                                                   (short)0, acc02, false, false);
    acc12 = __builtin_amdgcn_wmma_f32_16x16x32_f16(false, a1, false, vb2,
                                                   (short)0, acc12, false, false);
  }

  // ---- store, non-temporal (output is write-once, keep hsT in L2) --------
  // C/D layout: VGPR r -> M = g*8 + r, N = lane&15.
  #pragma unroll
  for (int r = 0; r < 8; ++r) {
    const int f_a = f0 + g * 8 + r;           // M-tile 0
    if (f_a < T_feats) {
      float* orow = out + ((size_t)b * T_feats + f_a) * CHN;
      __builtin_nontemporal_store(acc00[r], &orow[ct0 * 16 + n]);
      __builtin_nontemporal_store(acc01[r], &orow[ct1 * 16 + n]);
      __builtin_nontemporal_store(acc02[r], &orow[ct2 * 16 + n]);
    }
    const int f_b = f0 + 16 + g * 8 + r;      // M-tile 1
    if (f_b < T_feats) {
      float* orow = out + ((size_t)b * T_feats + f_b) * CHN;
      __builtin_nontemporal_store(acc10[r], &orow[ct0 * 16 + n]);
      __builtin_nontemporal_store(acc11[r], &orow[ct1 * 16 + n]);
      __builtin_nontemporal_store(acc12[r], &orow[ct2 * 16 + n]);
    }
  }
}

// ---------------------------------------------------------------------------
extern "C" void kernel_launch(void* const* d_in, const int* in_sizes, int n_in,
                              void* d_out, int out_size, void* d_ws, size_t ws_size,
                              hipStream_t stream) {
  const float*         hs = (const float*)d_in[0];
  const int*           ds = (const int*)d_in[1];
  const unsigned char* dm = (const unsigned char*)d_in[2];
  (void)in_sizes; (void)n_in; (void)ws_size;

  // T_feats is data-dependent on device, but out_size = B*T_feats*C (host).
  const int T_feats = out_size / (NB * CHN);

  float*    cbuf = (float*)d_ws;                                           // 16 KB
  _Float16* hsT  = (_Float16*)((char*)d_ws + NB * T_TEXT * sizeof(float)); // 3 MB

  gup_centers_kernel<<<NB, 32, 0, stream>>>(ds, cbuf);

  const int nel = NB * T_TEXT * CHN;
  gup_cvt_kernel<<<(nel + 255) / 256, 256, 0, stream>>>(hs, hsT);

  dim3 grid((T_feats + 31) / 32, NB);
  gup_main_kernel<<<grid, 256, 0, stream>>>(cbuf, dm, hsT, (float*)d_out, T_feats);
}
